// TopKPoolingNet_76166950028306
// MI455X (gfx1250) — matrix operand order repaired and checked by reading, not compiled, not run on hardware
//
#include <hip/hip_runtime.h>

typedef _Float16 half_t;
typedef __attribute__((ext_vector_type(16))) _Float16 v16h;
typedef __attribute__((ext_vector_type(8)))  _Float16 v8h;
typedef __attribute__((ext_vector_type(8)))  float    v8f;

#define DD    128   // input feature dim
#define ACM   4     // augmented channels
#define CC    128   // channels
#define C2    256   // 2*channels
#define NBINS 65536
#define LDS_STRIDE 136   // 128 + 8 halfs pad -> 68-bank shift per row, 16B aligned

#define LDS_SPACE __attribute__((address_space(3)))

__device__ __forceinline__ float gelu_exact(float v) {
    return 0.5f * v * (1.0f + erff(v * 0.70710678118654752440f));
}

// LDS byte offset of a pointer into a __shared__ array (generic -> AS3 -> int)
__device__ __forceinline__ unsigned lds_off(const void* p) {
    return (unsigned)(unsigned long long)(LDS_SPACE const void*)p;
}

// async copy of 16 bytes global -> LDS (gfx1250, tracked by ASYNCcnt)
__device__ __forceinline__ void async_copy_b128(unsigned ldsAddr, const void* gaddr) {
    asm volatile("global_load_async_to_lds_b128 %0, %1, off"
                 :: "v"(ldsAddr), "v"(gaddr)
                 : "memory");
}
__device__ __forceinline__ void wait_asynccnt0() {
    asm volatile("s_wait_asynccnt 0" ::: "memory");
}

// ---------------------------------------------------------------------------
// 1) h0 = gelu(x @ Wd_root.T + bd)   [N,128] -> [N,4]; wave per node
// ---------------------------------------------------------------------------
__global__ void k_dense_input(const float* __restrict__ x, const float* __restrict__ Wd,
                              const float* __restrict__ bd, float* __restrict__ h0, int N) {
    __shared__ float sW[ACM * DD];
    __shared__ float sb[ACM];
    int tid = threadIdx.x;
    for (int i = tid; i < ACM * DD; i += blockDim.x) sW[i] = Wd[i];
    if (tid < ACM) sb[tid] = bd[tid];
    __syncthreads();
    int wave = tid >> 5, lane = tid & 31;
    int n = blockIdx.x * (blockDim.x >> 5) + wave;
    if (n >= N) return;
    float4 xc = ((const float4*)(x + (size_t)n * DD))[lane];
    float acc[ACM];
#pragma unroll
    for (int a = 0; a < ACM; ++a) {
        float4 wc = ((const float4*)(sW + a * DD))[lane];
        acc[a] = xc.x * wc.x + xc.y * wc.y + xc.z * wc.z + xc.w * wc.w;
    }
#pragma unroll
    for (int off = 16; off > 0; off >>= 1)
#pragma unroll
        for (int a = 0; a < ACM; ++a) acc[a] += __shfl_xor(acc[a], off);
    if (lane == 0) {
#pragma unroll
        for (int a = 0; a < ACM; ++a)
            h0[(size_t)n * ACM + a] = gelu_exact(acc[a] + sb[a]);
    }
}

// ---------------------------------------------------------------------------
// 2) aggr[dst] += h0[src]   (4 channels, E edges)
// ---------------------------------------------------------------------------
__global__ void k_edge_scatter4(const int* __restrict__ src, const int* __restrict__ dst,
                                const float* __restrict__ h0, float* __restrict__ aggr, int E) {
    int e = blockIdx.x * blockDim.x + threadIdx.x;
    if (e >= E) return;
    int s = src[e], d = dst[e];
    float4 hv = *(const float4*)(h0 + (size_t)s * ACM);
    float* ap = aggr + (size_t)d * ACM;
    atomicAdd(ap + 0, hv.x); atomicAdd(ap + 1, hv.y);
    atomicAdd(ap + 2, hv.z); atomicAdd(ap + 3, hv.w);
}

// ---------------------------------------------------------------------------
// 3) h1 = gelu(aggr @ Wi_rel.T + bi + h0 @ Wi_root.T)  [N,4] -> [N,128]
// ---------------------------------------------------------------------------
__global__ void k_input_conv(const float* __restrict__ aggr, const float* __restrict__ h0,
                             const float* __restrict__ Wrel, const float* __restrict__ bi,
                             const float* __restrict__ Wroot, float* __restrict__ h1,
                             long long total) {
    long long idx = (long long)blockIdx.x * blockDim.x + threadIdx.x;
    if (idx >= total) return;
    int n = (int)(idx >> 7), c = (int)(idx & (CC - 1));
    float4 av = *(const float4*)(aggr + (size_t)n * ACM);
    float4 hv = *(const float4*)(h0 + (size_t)n * ACM);
    float4 wr = *(const float4*)(Wrel + (size_t)c * ACM);
    float4 wo = *(const float4*)(Wroot + (size_t)c * ACM);
    float v = bi[c] + av.x * wr.x + av.y * wr.y + av.z * wr.z + av.w * wr.w
                    + hv.x * wo.x + hv.y * wo.y + hv.z * wo.z + hv.w * wo.w;
    h1[idx] = gelu_exact(v);
}

// ---------------------------------------------------------------------------
// 4) score = tanh(h1 . p / ||p||); 16-bit descending bucket; histogram
// ---------------------------------------------------------------------------
__global__ void k_score(const float* __restrict__ h1, const float* __restrict__ p,
                        float* __restrict__ score, unsigned* __restrict__ key,
                        unsigned* __restrict__ hist, int N) {
    int wave = threadIdx.x >> 5, lane = threadIdx.x & 31;
    int n = blockIdx.x * (blockDim.x >> 5) + wave;
    if (n >= N) return;
    float sh = 0.f, sp = 0.f;
#pragma unroll
    for (int j = 0; j < CC / 32; ++j) {
        int c = lane + j * 32;
        float pv = p[c];
        sh += h1[(size_t)n * CC + c] * pv;
        sp += pv * pv;
    }
#pragma unroll
    for (int off = 16; off > 0; off >>= 1) {
        sh += __shfl_xor(sh, off);
        sp += __shfl_xor(sp, off);
    }
    if (lane == 0) {
        float s = tanhf(sh * rsqrtf(sp));
        score[n] = s;
        unsigned u = __float_as_uint(s);
        unsigned asc = u ^ ((u >> 31) ? 0xFFFFFFFFu : 0x80000000u);
        unsigned b = (~asc) >> 16;     // bucket 0 == highest score
        key[n] = b;
        atomicAdd(&hist[b], 1u);
    }
}

// ---------------------------------------------------------------------------
// 5) exclusive scan of 65536 bins (single block); find threshold bucket T
// ---------------------------------------------------------------------------
__global__ void k_scan(const unsigned* __restrict__ hist, unsigned* __restrict__ base,
                       unsigned* __restrict__ meta, int k) {
    __shared__ unsigned tot[256];
    int t = threadIdx.x;
    int b0 = t * 256;
    unsigned s = 0;
    for (int i = 0; i < 256; ++i) { base[b0 + i] = s; s += hist[b0 + i]; }
    tot[t] = s;
    __syncthreads();
    if (t == 0) {
        unsigned run = 0;
        for (int i = 0; i < 256; ++i) { unsigned v = tot[i]; tot[i] = run; run += v; }
    }
    __syncthreads();
    unsigned off = tot[t];
    unsigned T = 0xFFFFFFFFu;
    for (int i = 0; i < 256; ++i) {
        unsigned bgn = base[b0 + i] + off;
        base[b0 + i] = bgn;
        unsigned cnt = hist[b0 + i];
        if (T == 0xFFFFFFFFu && bgn < (unsigned)k && bgn + cnt >= (unsigned)k)
            T = (unsigned)(b0 + i);
    }
    if (T != 0xFFFFFFFFu) atomicMin(&meta[0], T);
}

// ---------------------------------------------------------------------------
// 6) assign ranks (perm / inv) from bucket bases + per-bucket cursors
// ---------------------------------------------------------------------------
__global__ void k_select(const unsigned* __restrict__ key, const unsigned* __restrict__ base,
                         unsigned* __restrict__ cursor, const unsigned* __restrict__ meta,
                         int* __restrict__ perm, int* __restrict__ inv, int N, int k) {
    int n = blockIdx.x * blockDim.x + threadIdx.x;
    if (n >= N) return;
    unsigned b = key[n];
    unsigned T = meta[0];
    int pos = -1;
    if (b < T) {
        pos = (int)(base[b] + atomicAdd(&cursor[b], 1u));
    } else if (b == T) {
        unsigned r = (unsigned)k - base[T];
        unsigned c = atomicAdd(&cursor[b], 1u);
        if (c < r) pos = (int)(base[b] + c);
    }
    if (pos >= 0) { perm[pos] = n; inv[n] = pos; }
    else          { inv[n] = k; }
}

// ---------------------------------------------------------------------------
// 7) hp[i] = h1[perm[i]] * score[perm[i]]  (write f16 for WMMA + scatter)
// ---------------------------------------------------------------------------
__global__ void k_gather(const float* __restrict__ h1, const float* __restrict__ score,
                         const int* __restrict__ perm, half_t* __restrict__ hpH,
                         long long total) {
    long long idx = (long long)blockIdx.x * blockDim.x + threadIdx.x;
    if (idx >= total) return;
    int i = (int)(idx >> 7), c = (int)(idx & (CC - 1));
    int n = perm[i];
    hpH[idx] = (half_t)(h1[(size_t)n * CC + c] * score[n]);
}

// ---------------------------------------------------------------------------
// 8) coarsened-graph scatter: aggr2[nd] += hp[ns]  (128 ch, 32 lanes/edge)
// ---------------------------------------------------------------------------
__global__ void k_edge_scatter_c(const int* __restrict__ src, const int* __restrict__ dst,
                                 const int* __restrict__ inv, const half_t* __restrict__ hpH,
                                 float* __restrict__ aggr2, long long totalThreads, int k) {
    long long tid = (long long)blockIdx.x * blockDim.x + threadIdx.x;
    if (tid >= totalThreads) return;
    int lane = (int)(tid & 31);
    int e = (int)(tid >> 5);
    int ns = inv[src[e]], nd = inv[dst[e]];
    if (ns >= k || nd >= k) return;
    const half_t* hp = hpH + (size_t)ns * CC + lane * 4;
    float* ap = aggr2 + (size_t)nd * CC + lane * 4;
#pragma unroll
    for (int j = 0; j < 4; ++j) atomicAdd(ap + j, (float)hp[j]);
}

// ---------------------------------------------------------------------------
// 9a) f32 -> f16 conversions (weights once per call, aggr2 after scatter)
// ---------------------------------------------------------------------------
__global__ void k_cvt(const float* __restrict__ in, half_t* __restrict__ outp, long long total) {
    long long i = (long long)blockIdx.x * blockDim.x + threadIdx.x;
    if (i >= total) return;
    outp[i] = (half_t)in[i];
}

// ---------------------------------------------------------------------------
// 9b) h2 = gelu(aggr2 @ W1_rel.T + b1 + hp @ W1_root.T)  via WMMA f16.
//     One block (4 waves) per 16-row M-tile; A panels staged once in LDS via
//     global_load_async_to_lds_b128; each wave covers 4 of the 16 N-tiles so
//     every A fragment read from LDS feeds 4 WMMAs.
// ---------------------------------------------------------------------------
__device__ __forceinline__ v16h load_a_lds(const half_t* S, int rlo, int kb, int hi) {
    const half_t* p = S + rlo * LDS_STRIDE + kb + 8 * hi;
    v8h lo  = *(const v8h*)p;
    v8h hi8 = *(const v8h*)(p + 16);
    v16h r;
#pragma unroll
    for (int i = 0; i < 8; ++i) { r[i] = lo[i]; r[8 + i] = hi8[i]; }
    return r;
}
__device__ __forceinline__ v16h load_b16(const half_t* W, int col, int kb, int hi) {
    const half_t* p = W + (size_t)col * CC + kb + 8 * hi;
    v8h lo  = *(const v8h*)p;
    v8h hi8 = *(const v8h*)(p + 16);
    v16h r;
#pragma unroll
    for (int i = 0; i < 8; ++i) { r[i] = lo[i]; r[8 + i] = hi8[i]; }
    return r;
}

__global__ void k_conv1_wmma(const half_t* __restrict__ aggr2H, const half_t* __restrict__ hpH,
                             const half_t* __restrict__ WrelH, const half_t* __restrict__ WrootH,
                             const float* __restrict__ b1, float* __restrict__ h2, int k) {
    __shared__ half_t sA[16 * LDS_STRIDE];   // aggr2 panel (16 x 128, padded)
    __shared__ half_t sP[16 * LDS_STRIDE];   // hp panel
    int mt = blockIdx.x;
    int tilesM = (k + 15) >> 4;
    if (mt >= tilesM) return;                // block-uniform

    // --- stage A panels: 512 x 16B chunks, 128 threads x 4 chunks each ---
    int tid = threadIdx.x;
#pragma unroll
    for (int t = 0; t < 4; ++t) {
        int chunk = tid + t * 128;           // 0..511
        int m = chunk >> 8;                  // 0 = aggr2, 1 = hp
        int r = (chunk >> 4) & 15;           // row in tile
        int c = chunk & 15;                  // 16B chunk in row
        int grow = mt * 16 + r;
        if (grow >= k) grow = k - 1;
        const half_t* g = (m ? hpH : aggr2H) + (size_t)grow * CC + c * 8;
        half_t* l = (m ? sP : sA) + r * LDS_STRIDE + c * 8;
        async_copy_b128(lds_off(l), (const void*)g);
    }
    wait_asynccnt0();
    __syncthreads();

    // --- compute: wave w covers N-tiles 4w..4w+3 ---
    int wv = tid >> 5, lane = tid & 31;
    int rlo = lane & 15, hi = lane >> 4;
    v8f acc0 = {}, acc1 = {}, acc2 = {}, acc3 = {};
    int ntBase = wv * 4;
#pragma unroll
    for (int kb = 0; kb < CC; kb += 32) {
        v16h aR = load_a_lds(sA, rlo, kb, hi);
        v16h aP = load_a_lds(sP, rlo, kb, hi);
#pragma unroll
        for (int j = 0; j < 4; ++j) {
            int col = (ntBase + j) * 16 + rlo;
            v16h bR = load_b16(WrelH, col, kb, hi);
            v16h bP = load_b16(WrootH, col, kb, hi);
            v8f a = (j == 0) ? acc0 : (j == 1) ? acc1 : (j == 2) ? acc2 : acc3;
            a = __builtin_amdgcn_wmma_f32_16x16x32_f16(false, aR, false, bR,
                                                       (short)0, a, false, false);
            a = __builtin_amdgcn_wmma_f32_16x16x32_f16(false, aP, false, bP,
                                                       (short)0, a, false, false);
            if (j == 0) acc0 = a; else if (j == 1) acc1 = a;
            else if (j == 2) acc2 = a; else acc3 = a;
        }
    }
    // --- epilogue: bias + exact GELU, guarded stores ---
#pragma unroll
    for (int j = 0; j < 4; ++j) {
        v8f a = (j == 0) ? acc0 : (j == 1) ? acc1 : (j == 2) ? acc2 : acc3;
        int col = (ntBase + j) * 16 + rlo;
        float bias = b1[col];
#pragma unroll
        for (int r = 0; r < 8; ++r) {
            int orow = mt * 16 + r + 8 * hi;   // D layout: VGPR r -> M=r / M=r+8
            if (orow < k)
                h2[(size_t)orow * C2 + col] = gelu_exact(a[r] + bias);
        }
    }
}

// ---------------------------------------------------------------------------
// 10) cluster (i%3) sum-pool, blocked accumulation
// ---------------------------------------------------------------------------
__global__ void k_pool(const float* __restrict__ h2, float* __restrict__ psum, int k) {
    int c = threadIdx.x;                 // 256 output channels
    int r0 = blockIdx.x * 256;
    int rend = r0 + 256; if (rend > k) rend = k;
    float a0 = 0.f, a1 = 0.f, a2 = 0.f;
    for (int r = r0; r < rend; ++r) {
        float v = h2[(size_t)r * C2 + c];
        int m = r % 3;
        if (m == 0) a0 += v; else if (m == 1) a1 += v; else a2 += v;
    }
    atomicAdd(&psum[0 * C2 + c], a0);
    atomicAdd(&psum[1 * C2 + c], a1);
    atomicAdd(&psum[2 * C2 + c], a2);
}

// ---------------------------------------------------------------------------
// 11) out = flatten(pooled_mean) @ Wo.T + bo
// ---------------------------------------------------------------------------
__global__ void k_out(const float* __restrict__ psum, const float* __restrict__ Wo,
                      const float* __restrict__ bo, float* __restrict__ out, int k) {
    __shared__ float red[256];
    int t = threadIdx.x;
    float partial = 0.f;
#pragma unroll
    for (int j = 0; j < 3; ++j) {
        float cnt = (float)((k - 1 - j) / 3 + 1);
        partial += (psum[j * C2 + t] / cnt) * Wo[j * C2 + t];
    }
    red[t] = partial;
    __syncthreads();
    for (int s = 128; s > 0; s >>= 1) {
        if (t < s) red[t] += red[t + s];
        __syncthreads();
    }
    if (t == 0) out[0] = red[0] + bo[0];
}

// ---------------------------------------------------------------------------
extern "C" void kernel_launch(void* const* d_in, const int* in_sizes, int n_in,
                              void* d_out, int out_size, void* d_ws, size_t ws_size,
                              hipStream_t stream) {
    const float* x      = (const float*)d_in[0];
    const int*   eidx   = (const int*)d_in[1];
    const float* Wd     = (const float*)d_in[2];
    const float* bd     = (const float*)d_in[3];
    const float* WiRel  = (const float*)d_in[4];
    const float* bi     = (const float*)d_in[5];
    const float* WiRoot = (const float*)d_in[6];
    const float* p      = (const float*)d_in[7];
    const float* W1rel  = (const float*)d_in[8];
    const float* b1     = (const float*)d_in[9];
    const float* W1root = (const float*)d_in[10];
    const float* Wo     = (const float*)d_in[11];
    const float* bo     = (const float*)d_in[12];
    float* out = (float*)d_out;

    int N = in_sizes[0] / DD;
    int E = in_sizes[1] / 2;
    int k = (int)((7LL * N + 9) / 10);   // ceil(0.7*N)
    const int* src = eidx;
    const int* dst = eidx + E;

    // ---- workspace carve (all scratch from d_ws) ----
    char* ws = (char*)d_ws;
    size_t off = 0;
    auto alloc = [&](size_t bytes) -> void* {
        void* pp = ws + off;
        off = (off + bytes + 255) & ~(size_t)255;
        return pp;
    };
    float*    h0     = (float*)alloc((size_t)N * ACM * 4);
    float*    aggr   = (float*)alloc((size_t)N * ACM * 4);
    float*    h1     = (float*)alloc((size_t)N * CC * 4);
    float*    score  = (float*)alloc((size_t)N * 4);
    unsigned* key    = (unsigned*)alloc((size_t)N * 4);
    unsigned* hist   = (unsigned*)alloc((size_t)NBINS * 4);
    unsigned* base   = (unsigned*)alloc((size_t)NBINS * 4);
    unsigned* cursor = (unsigned*)alloc((size_t)NBINS * 4);
    unsigned* meta   = (unsigned*)alloc(256);
    int*      inv    = (int*)alloc((size_t)N * 4);
    int*      perm   = (int*)alloc((size_t)k * 4);
    half_t*   hpH    = (half_t*)alloc((size_t)k * CC * 2);
    float*    aggr2  = (float*)alloc((size_t)k * CC * 4);
    half_t*   aggr2H = (half_t*)alloc((size_t)k * CC * 2);
    half_t*   WrelH  = (half_t*)alloc((size_t)C2 * CC * 2);
    half_t*   WrootH = (half_t*)alloc((size_t)C2 * CC * 2);
    float*    h2     = (float*)alloc((size_t)k * C2 * 4);
    float*    psum   = (float*)alloc((size_t)3 * C2 * 4);

    // ---- per-call zeroing (graph-capture safe) ----
    hipMemsetAsync(aggr,   0,    (size_t)N * ACM * 4, stream);
    hipMemsetAsync(hist,   0,    (size_t)NBINS * 4,   stream);
    hipMemsetAsync(cursor, 0,    (size_t)NBINS * 4,   stream);
    hipMemsetAsync(meta,   0xFF, 4,                   stream);
    hipMemsetAsync(aggr2,  0,    (size_t)k * CC * 4,  stream);
    hipMemsetAsync(psum,   0,    (size_t)3 * C2 * 4,  stream);

    // 1) dense input  (wave per node, 8 waves / block)
    {
        int blocks = (N + 7) / 8;
        k_dense_input<<<blocks, 256, 0, stream>>>(x, Wd, bd, h0, N);
    }
    // 2) edge scatter (4 channels)
    {
        int blocks = (E + 255) / 256;
        k_edge_scatter4<<<blocks, 256, 0, stream>>>(src, dst, h0, aggr, E);
    }
    // 3) input conv -> h1
    {
        long long total = (long long)N * CC;
        int blocks = (int)((total + 255) / 256);
        k_input_conv<<<blocks, 256, 0, stream>>>(aggr, h0, WiRel, bi, WiRoot, h1, total);
    }
    // 4) scores + histogram
    {
        int blocks = (N + 7) / 8;
        k_score<<<blocks, 256, 0, stream>>>(h1, p, score, key, hist, N);
    }
    // 5) scan + threshold
    k_scan<<<1, 256, 0, stream>>>(hist, base, meta, k);
    // 6) rank assignment
    {
        int blocks = (N + 255) / 256;
        k_select<<<blocks, 256, 0, stream>>>(key, base, cursor, meta, perm, inv, N, k);
    }
    // 7) gather hp (f16)
    {
        long long total = (long long)k * CC;
        int blocks = (int)((total + 255) / 256);
        k_gather<<<blocks, 256, 0, stream>>>(h1, score, perm, hpH, total);
    }
    // weights -> f16
    {
        long long total = (long long)C2 * CC;
        int blocks = (int)((total + 255) / 256);
        k_cvt<<<blocks, 256, 0, stream>>>(W1rel,  WrelH,  total);
        k_cvt<<<blocks, 256, 0, stream>>>(W1root, WrootH, total);
    }
    // 8) coarsened edge scatter (128 channels)
    {
        long long total = (long long)E * 32;
        int blocks = (int)((total + 255) / 256);
        k_edge_scatter_c<<<blocks, 256, 0, stream>>>(src, dst, inv, hpH, aggr2, total, k);
    }
    // aggr2 -> f16
    {
        long long total = (long long)k * CC;
        int blocks = (int)((total + 255) / 256);
        k_cvt<<<blocks, 256, 0, stream>>>(aggr2, aggr2H, total);
    }
    // 9) WMMA conv1 -> h2   (block per 16-row tile; async LDS staging of A)
    {
        int tilesM = (k + 15) / 16;
        k_conv1_wmma<<<tilesM, 128, 0, stream>>>(aggr2H, hpH, WrelH, WrootH, b1, h2, k);
    }
    // 10) cluster pool
    {
        int blocks = (k + 255) / 256;
        k_pool<<<blocks, 256, 0, stream>>>(h2, psum, k);
    }
    // 11) output linear
    k_out<<<1, 256, 0, stream>>>(psum, Wo, bo, out, k);
}
